// SwinGNN_36077725287097
// MI455X (gfx1250) — matrix operand order repaired and checked by reading, not compiled, not run on hardware
//
#include <hip/hip_runtime.h>

typedef __attribute__((ext_vector_type(16))) __bf16 v16bf;
typedef __attribute__((ext_vector_type(8)))  __bf16 v8bf;
typedef __attribute__((ext_vector_type(4)))  __bf16 v4bf;
typedef __attribute__((ext_vector_type(8)))  float  v8f;
typedef __attribute__((__vector_size__(16))) int    i32x4;

#define NODES 49
#define DIN   768
#define DH1   256
#define DH2   128

// Hardware bf16 convert (fptrunc -> v_cvt_*bf16* on gfx1250, RNE).
__device__ __forceinline__ __bf16 to_bf16(float f) { return (__bf16)f; }

__device__ __forceinline__ v8bf cvt_v8(v8f a) {
    v8bf r;
#pragma unroll
    for (int i = 0; i < 8; ++i) r[i] = (__bf16)a[i];
    return r;
}

__device__ __forceinline__ v8f vzero8() {
    v8f z;
#pragma unroll
    for (int i = 0; i < 8; ++i) z[i] = 0.0f;
    return z;
}

// Pure 16B-granular global->LDS copy. Prefer CDNA5 async load-to-LDS
// (GLOBAL_LOAD_ASYNC_TO_LDS_B128, ASYNCcnt) when the builtin exists.
__device__ __forceinline__ void copy_to_lds_16B(const void* gsrc, void* lds_dst,
                                                int n16, int tid) {
#if __has_builtin(__builtin_amdgcn_global_load_async_to_lds_b128)
    for (int i = tid; i < n16; i += 256)
        __builtin_amdgcn_global_load_async_to_lds_b128(
            (__attribute__((address_space(1))) i32x4*)((char*)gsrc + (size_t)i * 16),
            (__attribute__((address_space(3))) i32x4*)((char*)lds_dst + (size_t)i * 16),
            0, 0);
#if __has_builtin(__builtin_amdgcn_s_wait_asynccnt)
    __builtin_amdgcn_s_wait_asynccnt(0);
#else
    asm volatile("s_wait_asynccnt 0" ::: "memory");
#endif
#else
    const uint4* s = (const uint4*)gsrc;
    uint4* d = (uint4*)lds_dst;
    for (int i = tid; i < n16; i += 256) d[i] = s[i];
#endif
}

// A fragment (16x32 bf16, row-major in LDS, row stride = `stride` elements).
// Lane l: row = l&15; VGPRs 0-3 hold K = (l>>4)*8 + 0..7, VGPRs 4-7 hold K = 16 + (l>>4)*8 + 0..7.
__device__ __forceinline__ v16bf load_a_frag(const __bf16* base, int stride, int ml, int lh) {
    const __bf16* p = base + ml * stride + lh * 8;
    v8bf lo = *(const v8bf*)p;
    v8bf hi = *(const v8bf*)(p + 16);
    return __builtin_shufflevector(lo, hi, 0,1,2,3,4,5,6,7,8,9,10,11,12,13,14,15);
}

// B fragment (32x16 bf16). LDS holds the tile column-major: column n is
// `colstride` contiguous K-halves. Lane l: col = l&15, VGPRs 0-3 hold K = (l>>4)*8 + 0..7, VGPRs 4-7 hold K = 16 + (l>>4)*8 + 0..7.
__device__ __forceinline__ v16bf load_b_frag(const __bf16* colbase, int colstride, int ml, int lh) {
    const __bf16* p = colbase + ml * colstride + lh * 8;
    v8bf lo = *(const v8bf*)p;
    v8bf hi = *(const v8bf*)(p + 16);
    return __builtin_shufflevector(lo, hi, 0,1,2,3,4,5,6,7,8,9,10,11,12,13,14,15);
}

__device__ __forceinline__ v8f wmma_bf16(v16bf a, v16bf b, v8f c) {
    return __builtin_amdgcn_wmma_f32_16x16x32_bf16(false, a, false, b, (short)0, c, false, false);
}

// ---------------------------------------------------------------------------
// Kernel 1: build normalized adjacency A_hat (49x49, padded to 64x64 w/ zeros), bf16.
// ---------------------------------------------------------------------------
__global__ __launch_bounds__(256) void build_ahat(const int* __restrict__ ei, int E,
                                                  __bf16* __restrict__ out) {
    __shared__ float A[64 * 64];
    __shared__ float dinv[64];
    const int tid = threadIdx.x;
    for (int i = tid; i < 64 * 64; i += 256) A[i] = 0.0f;
    __syncthreads();
    if (tid == 0) {
        for (int e = 0; e < E; ++e) {
            int s = ei[e], d = ei[E + e];
            A[d * 64 + s] += 1.0f;
        }
        for (int i = 0; i < NODES; ++i) A[i * 64 + i] += 1.0f;  // GCNConv's extra self-loop
    }
    __syncthreads();
    if (tid < 64) {
        float deg = 0.0f;
        for (int n = 0; n < NODES; ++n) deg += A[tid * 64 + n];
        dinv[tid] = (tid < NODES && deg > 0.0f) ? rsqrtf(deg) : 0.0f;
    }
    __syncthreads();
    for (int i = tid; i < 64 * 64; i += 256) {
        int m = i >> 6, n = i & 63;
        out[i] = to_bf16(A[i] * dinv[m] * dinv[n]);
    }
}

// ---------------------------------------------------------------------------
// Kernel 2: pack W1 [768,256] and W2 [256,128] -> bf16, transposed, K-chunk-major:
//   w1p[kc][n][k]  (kc = 24 chunks of K=32, n = 256, k = 32)
//   w2p[kc][n][k]  (kc = 8  chunks of K=32, n = 128, k = 32)
// ---------------------------------------------------------------------------
__global__ __launch_bounds__(256) void pack_weights(const float* __restrict__ W1,
                                                    const float* __restrict__ W2,
                                                    __bf16* __restrict__ w1p,
                                                    __bf16* __restrict__ w2p) {
    int i = blockIdx.x * 256 + threadIdx.x;
    if (i < 24 * 256 * 32) {
        int kc = i >> 13, r = i & 8191, n = r >> 5, k = r & 31;
        w1p[i] = to_bf16(W1[(kc * 32 + k) * DH1 + n]);
    }
    if (i < 8 * 128 * 32) {
        int kc = i >> 12, r = i & 4095, n = r >> 5, k = r & 31;
        w2p[i] = to_bf16(W2[(kc * 32 + k) * DH2 + n]);
    }
}

// ---------------------------------------------------------------------------
// Kernel 3: fully fused GCN for one batch per block (256 threads = 8 waves).
// ---------------------------------------------------------------------------
__global__ __launch_bounds__(256) void gcn_fused(const float* __restrict__ X,
                                                 const float* __restrict__ b1,
                                                 const float* __restrict__ b2,
                                                 const float* __restrict__ fcW,
                                                 const float* __restrict__ fcb,
                                                 const __bf16* __restrict__ ahat,
                                                 const __bf16* __restrict__ w1p,
                                                 const __bf16* __restrict__ w2p,
                                                 float* __restrict__ out) {
    __shared__ __align__(16) __bf16 s_ahat[64 * 64];   //  8 KB: A_hat padded
    __shared__ __align__(16) __bf16 s_a[64 * 32];      //  4 KB: X K-chunk
    __shared__ __align__(16) __bf16 s_w[256 * 32];     // 16 KB: weight K-chunk (col-major)
    __shared__ __align__(16) __bf16 s_cm[64 * 256];    // 32 KB: staging (Y1 cm / H1 rm / Y2 cm)
    __shared__ float s_pool[256];
    __shared__ float s_pooled[128];

    const int tid  = threadIdx.x;
    const int lane = tid & 31;
    const int wave = tid >> 5;        // 8 waves
    const int ml   = lane & 15;       // row/col within 16
    const int lh   = lane >> 4;       // lane-half select
    const int b    = blockIdx.x;

    // preload A_hat (8 KB)
    copy_to_lds_16B(ahat, s_ahat, 512, tid);

    const float* Xb = X + (size_t)b * (NODES * DIN);

    // ================= layer 1 GEMM: Y1 = X @ W1  (64x256, K=768) ================
    // wave w owns output columns [32w, 32w+32): 2 N-tiles x 4 M-tiles.
    v8f acc[2][4];
#pragma unroll
    for (int nt = 0; nt < 2; ++nt)
#pragma unroll
        for (int t = 0; t < 4; ++t) acc[nt][t] = vzero8();

    {
        // prefetch first X chunk
        int r0 = tid >> 3, c0 = (tid & 7) * 4;
        if (r0 < NODES) __builtin_prefetch(Xb + r0 * DIN + c0, 0, 0);
    }

    for (int kc = 0; kc < 24; ++kc) {
        __syncthreads();
        // X chunk -> s_a [64 rows x 32 K] bf16, rows >= 49 zeroed
        {
            int r0 = tid >> 3;             // 0..31
            int c0 = (tid & 7) * 4;        // 0,4,...,28
#pragma unroll
            for (int p = 0; p < 2; ++p) {
                int r = r0 + p * 32;
                float4 v = make_float4(0.f, 0.f, 0.f, 0.f);
                if (r < NODES) {
                    v = *(const float4*)(Xb + r * DIN + kc * 32 + c0);
                    if (kc + 1 < 24)
                        __builtin_prefetch(Xb + r * DIN + (kc + 1) * 32 + c0, 0, 0);
                }
                v4bf h;
                h[0] = to_bf16(v.x); h[1] = to_bf16(v.y);
                h[2] = to_bf16(v.z); h[3] = to_bf16(v.w);
                *(v4bf*)(s_a + r * 32 + c0) = h;
            }
        }
        // W1 chunk (16 KB contiguous) -> s_w
        copy_to_lds_16B(w1p + kc * 8192, s_w, 1024, tid);
        __syncthreads();

        v16bf bf0 = load_b_frag(s_w + (wave * 32 + 0) * 32, 32, ml, lh);
        v16bf bf1 = load_b_frag(s_w + (wave * 32 + 16) * 32, 32, ml, lh);
#pragma unroll
        for (int t = 0; t < 4; ++t) {
            v16bf af = load_a_frag(s_a + t * 16 * 32, 32, ml, lh);
            acc[0][t] = wmma_bf16(af, bf0, acc[0][t]);
            acc[1][t] = wmma_bf16(af, bf1, acc[1][t]);
        }
    }

    // stage Y1 column-major into s_cm (bf16): s_cm[c*64 + m]; 8 contiguous halves per tile
    __syncthreads();
#pragma unroll
    for (int nt = 0; nt < 2; ++nt) {
        int c = wave * 32 + nt * 16 + ml;
#pragma unroll
        for (int t = 0; t < 4; ++t)
            *(v8bf*)(s_cm + c * 64 + t * 16 + 8 * lh) = cvt_v8(acc[nt][t]);
    }
    __syncthreads();

    // ============ aggregation 1: H1 = relu(A_hat @ Y1 + b1)  (K = 64) ============
    v8f agg[2][4];
#pragma unroll
    for (int nt = 0; nt < 2; ++nt)
#pragma unroll
        for (int t = 0; t < 4; ++t) agg[nt][t] = vzero8();
#pragma unroll
    for (int s = 0; s < 2; ++s) {
        v16bf bf0 = load_b_frag(s_cm + (wave * 32 + 0) * 64 + s * 32, 64, ml, lh);
        v16bf bf1 = load_b_frag(s_cm + (wave * 32 + 16) * 64 + s * 32, 64, ml, lh);
#pragma unroll
        for (int t = 0; t < 4; ++t) {
            v16bf af = load_a_frag(s_ahat + t * 16 * 64 + s * 32, 64, ml, lh);
            agg[0][t] = wmma_bf16(af, bf0, agg[0][t]);
            agg[1][t] = wmma_bf16(af, bf1, agg[1][t]);
        }
    }
    __syncthreads();   // everyone done reading s_cm(Y1) before overwrite

    // bias + relu, restage H1 row-major into s_cm: s_cm[m*256 + c]
#pragma unroll
    for (int nt = 0; nt < 2; ++nt) {
        int c = wave * 32 + nt * 16 + ml;
        float bb = b1[c];
#pragma unroll
        for (int t = 0; t < 4; ++t)
#pragma unroll
            for (int v = 0; v < 8; ++v) {
                int m = t * 16 + v + 8 * lh;
                s_cm[m * 256 + c] = to_bf16(fmaxf(agg[nt][t][v] + bb, 0.0f));
            }
    }
    __syncthreads();

    // ================= layer 2 GEMM: Y2 = H1 @ W2  (64x128, K=256) ================
    // wave w owns output columns [16w, 16w+16).
    v8f acc2[4];
#pragma unroll
    for (int t = 0; t < 4; ++t) acc2[t] = vzero8();
    for (int kc = 0; kc < 8; ++kc) {
        __syncthreads();
        copy_to_lds_16B(w2p + kc * 4096, s_w, 512, tid);
        __syncthreads();
        v16bf bf0 = load_b_frag(s_w + (wave * 16) * 32, 32, ml, lh);
#pragma unroll
        for (int t = 0; t < 4; ++t) {
            v16bf af = load_a_frag(s_cm + t * 16 * 256 + kc * 32, 256, ml, lh);
            acc2[t] = wmma_bf16(af, bf0, acc2[t]);
        }
    }
    __syncthreads();   // done reading s_cm(H1)

    // stage Y2 column-major into s_cm
    {
        int c = wave * 16 + ml;
#pragma unroll
        for (int t = 0; t < 4; ++t)
            *(v8bf*)(s_cm + c * 64 + t * 16 + 8 * lh) = cvt_v8(acc2[t]);
    }
    __syncthreads();

    // ============ aggregation 2 + relu + masked mean pool over nodes ============
    v8f agg2[4];
#pragma unroll
    for (int t = 0; t < 4; ++t) agg2[t] = vzero8();
#pragma unroll
    for (int s = 0; s < 2; ++s) {
        v16bf bf0 = load_b_frag(s_cm + (wave * 16) * 64 + s * 32, 64, ml, lh);
#pragma unroll
        for (int t = 0; t < 4; ++t) {
            v16bf af = load_a_frag(s_ahat + t * 16 * 64 + s * 32, 64, ml, lh);
            agg2[t] = wmma_bf16(af, bf0, agg2[t]);
        }
    }
    {
        int c = wave * 16 + ml;
        float bb = b2[c];
        float s = 0.0f;
#pragma unroll
        for (int t = 0; t < 4; ++t)
#pragma unroll
            for (int v = 0; v < 8; ++v) {
                int m = t * 16 + v + 8 * lh;
                float h = fmaxf(agg2[t][v] + bb, 0.0f);
                s += (m < NODES) ? h : 0.0f;
            }
        s_pool[c * 2 + lh] = s;
    }
    __syncthreads();
    if (tid < 128)
        s_pooled[tid] = (s_pool[2 * tid] + s_pool[2 * tid + 1]) * (1.0f / (float)NODES);
    __syncthreads();

    // final FC: [128] -> [4]
    if (tid < 4) {
        float o = fcb[tid];
        for (int c = 0; c < 128; ++c) o += s_pooled[c] * fcW[c * 4 + tid];
        out[(size_t)b * 4 + tid] = o;
    }
}

// ---------------------------------------------------------------------------
extern "C" void kernel_launch(void* const* d_in, const int* in_sizes, int n_in,
                              void* d_out, int out_size, void* d_ws, size_t ws_size,
                              hipStream_t stream) {
    const float* X   = (const float*)d_in[0];
    const float* W1  = (const float*)d_in[1];
    const float* b1  = (const float*)d_in[2];
    const float* W2  = (const float*)d_in[3];
    const float* b2  = (const float*)d_in[4];
    const float* fcW = (const float*)d_in[5];
    const float* fcb = (const float*)d_in[6];
    const int*   ei  = (const int*)d_in[7];

    const int E = in_sizes[7] / 2;
    const int B = in_sizes[0] / (NODES * DIN);

    char* ws = (char*)d_ws;
    __bf16* ahat = (__bf16*)ws;                                   //   8 KB
    __bf16* w1p  = (__bf16*)(ws + 8192);                          // 384 KB
    __bf16* w2p  = (__bf16*)(ws + 8192 + 24 * 256 * 32 * 2);      //  64 KB

    build_ahat<<<1, 256, 0, stream>>>(ei, E, ahat);
    pack_weights<<<(24 * 256 * 32 + 255) / 256, 256, 0, stream>>>(W1, W2, w1p, w2p);
    gcn_fused<<<B, 256, 0, stream>>>(X, b1, b2, fcW, fcb, ahat, w1p, w2p, (float*)d_out);
}
